// MultiHeadAttentionBlock_58454504899095
// MI455X (gfx1250) — compile-verified, matrix operand order checked by reading, not run on hardware
//
#include <hip/hip_runtime.h>
#include <stdint.h>

#define D_MODEL 1024
#define NH 16
#define DK 64
#define BB 2
#define SS 2048

typedef __bf16 bf16x16 __attribute__((ext_vector_type(16)));
typedef float f32x8 __attribute__((ext_vector_type(8)));
typedef int v4i __attribute__((ext_vector_type(4)));
typedef unsigned int u32x4 __attribute__((ext_vector_type(4)));
typedef int i32x8 __attribute__((ext_vector_type(8)));

#if __has_builtin(__builtin_amdgcn_tensor_load_to_lds) && \
    __has_builtin(__builtin_amdgcn_s_wait_tensorcnt)
#define HAS_TDM 1
#else
#define HAS_TDM 0
#endif

#if __has_builtin(__builtin_amdgcn_global_load_async_to_lds_b128)
#define HAS_ASYNC 1
#else
#define HAS_ASYNC 0
#endif

#if HAS_ASYNC
#if __has_builtin(__builtin_amdgcn_s_wait_asynccnt)
#define WAIT_ASYNC(n) __builtin_amdgcn_s_wait_asynccnt(n)
#else
#define WAIT_ASYNC(n) asm volatile("s_wait_asynccnt %0" ::"i"(n) : "memory")
#endif
#else
#define WAIT_ASYNC(n) ((void)0)
#endif

union ABFrag { bf16x16 v; uint32_t u[8]; };

// copy 16 bytes global -> LDS (async on gfx1250, sync fallback elsewhere)
__device__ inline void async_cp16(const void* g, void* l) {
#if HAS_ASYNC
  __builtin_amdgcn_global_load_async_to_lds_b128(
      (__attribute__((address_space(1))) v4i*)(void*)g,
      (__attribute__((address_space(3))) v4i*)l, 0, 0);
#else
  *(uint4*)l = *(const uint4*)g;
#endif
}

#if HAS_TDM
// TDM 2D tile load: tile_d1 rows x tile_d0 elems (bf16), row stride stride0, packed into LDS.
// D# bit layout per CDNA5 ISA §8.3/8.4 (group0: count/lds/global/type, group1: dims/tile/stride).
__device__ inline void tdm_load_2d(const void* gaddr, void* lds, uint32_t tensor_d0,
                                   uint32_t tensor_d1, uint32_t tile_d0, uint32_t tile_d1,
                                   uint32_t stride0) {
  uint64_t ga = (uint64_t)(uintptr_t)gaddr;
  uint32_t la = (uint32_t)(uintptr_t)(__attribute__((address_space(3))) void*)lds;
  u32x4 g0;
  g0[0] = 1u;                                               // count=1, user mode
  g0[1] = la;                                               // lds_addr (bytes)
  g0[2] = (uint32_t)ga;                                     // global_addr[31:0]
  g0[3] = (uint32_t)((ga >> 32) & 0x01FFFFFFu) | (2u << 30);// global_addr[56:32], type=2
  i32x8 g1;
  g1[0] = (int)(1u << 16);                                  // workgroup_mask=0, data_size=1 (2B)
  g1[1] = (int)(tensor_d0 << 16);                           // tensor_dim0[15:0]
  g1[2] = (int)((tensor_d0 >> 16) | (tensor_d1 << 16));     // tensor_dim0[31:16]|tensor_dim1[15:0]
  g1[3] = (int)((tensor_d1 >> 16) | (tile_d0 << 16));       // tensor_dim1[31:16]|tile_dim0
  g1[4] = (int)(tile_d1 & 0xFFFFu);                         // tile_dim1, tile_dim2=0
  g1[5] = (int)stride0;                                     // tensor_dim0_stride[31:0]
  g1[6] = 0;                                                // stride0[47:32], dim1_stride lo
  g1[7] = 0;
  v4i z4 = {0, 0, 0, 0};                                    // groups 2/3 unused (2D tensor)
  i32x8 z8 = {0, 0, 0, 0, 0, 0, 0, 0};
  __builtin_amdgcn_tensor_load_to_lds(g0, g1, z4, z4, z8, 0);
}
#endif

__device__ inline uint16_t f32_bf16(float f) {
  union { float f; uint32_t u; } c; c.f = f;
  uint32_t u = c.u;
  return (uint16_t)((u + 0x7FFFu + ((u >> 16) & 1u)) >> 16);
}
__device__ inline uint32_t pack_bf16x2(float lo, float hi) {
  return (uint32_t)f32_bf16(lo) | ((uint32_t)f32_bf16(hi) << 16);
}
// ISA 16-bit A/B fragment layout: pair p (0..7), lane-half hi -> k of the even element
__device__ inline int kpair(int p, int hi) {
  return ((p & 4) << 2) + ((p & 3) << 1) + (hi << 3);
}

// Y[4096,1024] = X[4096,1024] @ W^T + bias
// MODE 0: X fp32, Y bf16 head-major [B,H,S,DK]; optional transposed copy Yt [B,H,DK,S]
// MODE 1: X bf16, Y fp32 row-major (final output)
template <int MODE>
__global__ __launch_bounds__(256) void gemm_xwt(const void* __restrict__ Xin,
                                                const float* __restrict__ W,
                                                const float* __restrict__ bias,
                                                void* __restrict__ Yout,
                                                uint16_t* __restrict__ Yt) {
  __shared__ uint16_t Xs[128 * 32];
  __shared__ uint16_t Ws[64 * 32];
  const int t = threadIdx.x;
  const int lane = t & 31;
  const int wave = t >> 5;
  const int hi = lane >> 4;
  const int col = lane & 15;
  const int m0wg = blockIdx.x * 128;
  const int n0wg = blockIdx.y * 64;

  f32x8 acc[4] = {};

  for (int k0 = 0; k0 < 1024; k0 += 32) {
    __syncthreads();
    {  // X tile -> LDS (bf16)
      int row = t >> 1;
      int c0 = (t & 1) * 16;
      int grow = m0wg + row;
      if (MODE == 0) {
        const float* src = (const float*)Xin + (size_t)grow * 1024 + k0 + c0;
        float4 a0 = *(const float4*)(src + 0);
        float4 a1 = *(const float4*)(src + 4);
        float4 a2 = *(const float4*)(src + 8);
        float4 a3 = *(const float4*)(src + 12);
        uint32_t* dst = (uint32_t*)&Xs[row * 32 + c0];
        *(uint4*)(dst + 0) = make_uint4(pack_bf16x2(a0.x, a0.y), pack_bf16x2(a0.z, a0.w),
                                        pack_bf16x2(a1.x, a1.y), pack_bf16x2(a1.z, a1.w));
        *(uint4*)(dst + 4) = make_uint4(pack_bf16x2(a2.x, a2.y), pack_bf16x2(a2.z, a2.w),
                                        pack_bf16x2(a3.x, a3.y), pack_bf16x2(a3.z, a3.w));
      } else {
        const uint16_t* src = (const uint16_t*)Xin + (size_t)grow * 1024 + k0 + c0;
        uint4* dst = (uint4*)&Xs[row * 32 + c0];
        dst[0] = ((const uint4*)src)[0];
        dst[1] = ((const uint4*)src)[1];
      }
    }
    {  // W tile -> LDS (bf16), rows are output cols n
      int row = t >> 2;
      int c0 = (t & 3) * 8;
      const float* src = W + (size_t)(n0wg + row) * 1024 + k0 + c0;
      float4 a0 = *(const float4*)(src + 0);
      float4 a1 = *(const float4*)(src + 4);
      *(uint4*)&Ws[row * 32 + c0] = make_uint4(pack_bf16x2(a0.x, a0.y), pack_bf16x2(a0.z, a0.w),
                                               pack_bf16x2(a1.x, a1.y), pack_bf16x2(a1.z, a1.w));
    }
    __syncthreads();

    ABFrag a;
    {
      int m = wave * 16 + col;
#pragma unroll
      for (int p = 0; p < 8; ++p) a.u[p] = *(const uint32_t*)&Xs[m * 32 + kpair(p, hi)];
    }
#pragma unroll
    for (int nt = 0; nt < 4; ++nt) {
      ABFrag b;
      int n = nt * 16 + col;
#pragma unroll
      for (int p = 0; p < 8; ++p) b.u[p] = *(const uint32_t*)&Ws[n * 32 + kpair(p, hi)];
      acc[nt] = __builtin_amdgcn_wmma_f32_16x16x32_bf16(false, a.v, false, b.v, (short)0,
                                                        acc[nt], false, false);
    }
  }

#pragma unroll
  for (int nt = 0; nt < 4; ++nt) {
#pragma unroll
    for (int i = 0; i < 8; ++i) {
      int r = m0wg + wave * 16 + i + hi * 8;
      int c = n0wg + nt * 16 + col;
      float v = acc[nt][i] + bias[c];
      if (MODE == 0) {
        int b_ = r >> 11, s_ = r & 2047;
        int h_ = c >> 6, d_ = c & 63;
        uint16_t bv = f32_bf16(v);
        ((uint16_t*)Yout)[(((size_t)(b_ * NH + h_) * SS) + s_) * DK + d_] = bv;
        if (Yt) Yt[((size_t)(b_ * NH + h_) * DK + d_) * SS + s_] = bv;
      } else {
        ((float*)Yout)[(size_t)r * 1024 + c] = v;
      }
    }
  }
}

// Flash attention per (b,h): scores = Q K^T / 8, mask==0 -> -1e9, softmax, out = P @ Q (V==Q bug)
// Qp: [B*H, S, DK] bf16 (A-side), Qt: [B*H, DK, S] bf16 (V^T tiles), Kp: [B*H, S, DK]
__global__ __launch_bounds__(256) void attn_kernel(const uint16_t* __restrict__ Qp,
                                                   const uint16_t* __restrict__ Qt,
                                                   const uint16_t* __restrict__ Kp,
                                                   const int* __restrict__ mask,
                                                   uint16_t* __restrict__ Xo) {
  __shared__ uint16_t Qs[128 * 64];
  __shared__ uint16_t Ks[2][64 * 64];   // [kv][dk], double buffered
  __shared__ uint16_t Vs[2][64 * 64];   // [dk][kv], double buffered
  __shared__ uint16_t Ps[8 * 16 * 64];  // per-wave P scratch

  const int t = threadIdx.x;
  const int lane = t & 31;
  const int wave = t >> 5;
  const int hi = lane >> 4;
  const int col = lane & 15;
  const int tile = blockIdx.x;  // 16 row tiles
  const int bh = blockIdx.y;    // 32 (b,h) pairs
  const int b = bh >> 4;
  const int h = bh & 15;

  const uint16_t* Qbh = Qp + (size_t)bh * SS * DK;
  const uint16_t* Kbh = Kp + (size_t)bh * SS * DK;
  const uint16_t* Vt = Qt + (size_t)bh * DK * SS;  // [dk][s]
  const int* maskB = mask + (size_t)b * SS * SS;

  // per-thread async-copy slice of one 64x64 tile pair (fallback path)
  const int prow = t >> 2;
  const int pc0 = (t & 3) * 16;

  // kick off prefetch of KV tile 0 into buffer 0 before staging Q
#if HAS_TDM
  if (wave == 0) {
    tdm_load_2d(Kbh, &Ks[0][0], DK, 64, DK, 64, DK);          // K tile: 64 rows of 64, stride 64
    tdm_load_2d(Vt, &Vs[0][0], 64, DK, 64, DK, SS);           // V^T tile: 64 rows of 64, stride S
  }
#else
  {
    const uint16_t* gk = Kbh + (size_t)prow * DK + pc0;
    async_cp16(gk, &Ks[0][prow * 64 + pc0]);
    async_cp16(gk + 8, &Ks[0][prow * 64 + pc0 + 8]);
    const uint16_t* gv = Vt + (size_t)prow * SS + pc0;
    async_cp16(gv, &Vs[0][prow * 64 + pc0]);
    async_cp16(gv + 8, &Vs[0][prow * 64 + pc0 + 8]);
  }
#endif

  {  // Q tile 128x64 -> LDS
    int row = t >> 1;
    int half = t & 1;
    const uint4* src = (const uint4*)(Qbh + (size_t)(tile * 128 + row) * DK + half * 32);
    uint4* dst = (uint4*)&Qs[row * 64 + half * 32];
    dst[0] = src[0]; dst[1] = src[1]; dst[2] = src[2]; dst[3] = src[3];
  }
  __syncthreads();

  ABFrag aQ[2];
  {
    int m = wave * 16 + col;
#pragma unroll
    for (int c = 0; c < 2; ++c)
#pragma unroll
      for (int p = 0; p < 8; ++p)
        aQ[c].u[p] = *(const uint32_t*)&Qs[m * 64 + kpair(p, hi) + c * 32];
  }

  f32x8 acc[4] = {};
  float m_run[8], l_run[8];
#pragma unroll
  for (int i = 0; i < 8; ++i) { m_run[i] = -3.0e38f; l_run[i] = 0.0f; }

  uint16_t* Pw = &Ps[wave * 16 * 64];

  for (int kv0 = 0; kv0 < SS; kv0 += 64) {
    const int cur = (kv0 >> 6) & 1;
    __syncthreads();  // all waves done reading buffer cur^1 (previous compute)
#if HAS_TDM
    if (wave == 0) {
      if (kv0 + 64 < SS) {  // prefetch next KV tile via Tensor Data Mover
        const int nxt = cur ^ 1;
        tdm_load_2d(Kbh + (size_t)(kv0 + 64) * DK, &Ks[nxt][0], DK, 64, DK, 64, DK);
        tdm_load_2d(Vt + (kv0 + 64), &Vs[nxt][0], 64, DK, 64, DK, SS);
        __builtin_amdgcn_s_wait_tensorcnt(2);  // in-order: the 2 older DMAs (cur) landed
      } else {
        __builtin_amdgcn_s_wait_tensorcnt(0);
      }
    }
#else
    if (kv0 + 64 < SS) {  // prefetch next tile into the other buffer
      const int nxt = cur ^ 1;
      const uint16_t* gk = Kbh + (size_t)(kv0 + 64 + prow) * DK + pc0;
      async_cp16(gk, &Ks[nxt][prow * 64 + pc0]);
      async_cp16(gk + 8, &Ks[nxt][prow * 64 + pc0 + 8]);
      const uint16_t* gv = Vt + (size_t)prow * SS + kv0 + 64 + pc0;
      async_cp16(gv, &Vs[nxt][prow * 64 + pc0]);
      async_cp16(gv + 8, &Vs[nxt][prow * 64 + pc0 + 8]);
      WAIT_ASYNC(4);  // in-order completion: the 4 older ops (buffer cur) have landed
    } else {
      WAIT_ASYNC(0);
    }
#endif
    __syncthreads();  // buffer cur is in LDS and visible to all waves

    f32x8 sc[4] = {};
#pragma unroll
    for (int c = 0; c < 2; ++c) {
#pragma unroll
      for (int nt = 0; nt < 4; ++nt) {
        ABFrag bk;
        int tc = nt * 16 + col;
#pragma unroll
        for (int p = 0; p < 8; ++p)
          bk.u[p] = *(const uint32_t*)&Ks[cur][tc * 64 + kpair(p, hi) + c * 32];
        sc[nt] = __builtin_amdgcn_wmma_f32_16x16x32_bf16(false, aQ[c].v, false, bk.v, (short)0,
                                                         sc[nt], false, false);
      }
    }
    // scale + mask (C layout: row = i + 8*hi, col = lane&15)
#pragma unroll
    for (int nt = 0; nt < 4; ++nt) {
#pragma unroll
      for (int i = 0; i < 8; ++i) {
        int sq = tile * 128 + wave * 16 + i + hi * 8;
        int tk = kv0 + nt * 16 + col;
        float v = sc[nt][i] * 0.125f;
        if (maskB[(size_t)sq * SS + tk] == 0) v = -1.0e9f;
        sc[nt][i] = v;
      }
    }
    // online softmax: row max/sum via 16-lane butterflies
#pragma unroll
    for (int i = 0; i < 8; ++i) {
      float v = fmaxf(fmaxf(sc[0][i], sc[1][i]), fmaxf(sc[2][i], sc[3][i]));
      v = fmaxf(v, __shfl_xor(v, 1, 32));
      v = fmaxf(v, __shfl_xor(v, 2, 32));
      v = fmaxf(v, __shfl_xor(v, 4, 32));
      v = fmaxf(v, __shfl_xor(v, 8, 32));
      float mn = fmaxf(m_run[i], v);
      float corr = __expf(m_run[i] - mn);
      m_run[i] = mn;
      l_run[i] *= corr;
#pragma unroll
      for (int nt = 0; nt < 4; ++nt) acc[nt][i] *= corr;
    }
#pragma unroll
    for (int i = 0; i < 8; ++i) {
      float rs = 0.0f;
#pragma unroll
      for (int nt = 0; nt < 4; ++nt) {
        float p = __expf(sc[nt][i] - m_run[i]);
        sc[nt][i] = p;
        rs += p;
      }
      rs += __shfl_xor(rs, 1, 32);
      rs += __shfl_xor(rs, 2, 32);
      rs += __shfl_xor(rs, 4, 32);
      rs += __shfl_xor(rs, 8, 32);
      l_run[i] += rs;
    }
    // P (C layout f32) -> per-wave LDS as bf16, reload as A fragments
#pragma unroll
    for (int nt = 0; nt < 4; ++nt)
#pragma unroll
      for (int i = 0; i < 8; ++i)
        Pw[(i + hi * 8) * 64 + nt * 16 + col] = f32_bf16(sc[nt][i]);
    asm volatile("s_wait_dscnt 0" ::: "memory");
#pragma unroll
    for (int c = 0; c < 2; ++c) {
      ABFrag ap;
#pragma unroll
      for (int p = 0; p < 8; ++p)
        ap.u[p] = *(const uint32_t*)&Pw[col * 64 + kpair(p, hi) + c * 32];
#pragma unroll
      for (int nt = 0; nt < 4; ++nt) {
        ABFrag bv;
        int dn = nt * 16 + col;
#pragma unroll
        for (int p = 0; p < 8; ++p)
          bv.u[p] = *(const uint32_t*)&Vs[cur][dn * 64 + kpair(p, hi) + c * 32];
        acc[nt] = __builtin_amdgcn_wmma_f32_16x16x32_bf16(false, ap.v, false, bv.v, (short)0,
                                                          acc[nt], false, false);
      }
    }
  }

#pragma unroll
  for (int nt = 0; nt < 4; ++nt) {
#pragma unroll
    for (int i = 0; i < 8; ++i) {
      int sq = tile * 128 + wave * 16 + i + hi * 8;
      int dn = nt * 16 + col;
      float o = acc[nt][i] / l_run[i];
      Xo[((size_t)b * SS + sq) * D_MODEL + h * DK + dn] = f32_bf16(o);
    }
  }
}

extern "C" void kernel_launch(void* const* d_in, const int* in_sizes, int n_in,
                              void* d_out, int out_size, void* d_ws, size_t ws_size,
                              hipStream_t stream) {
  const float* q = (const float*)d_in[0];
  const float* k = (const float*)d_in[1];
  // d_in[2] (v), d_in[8] (Wv), d_in[9] (bv): dead in the reference (value = query bug)
  const int* mask = (const int*)d_in[3];
  const float* Wq = (const float*)d_in[4];
  const float* bq = (const float*)d_in[5];
  const float* Wk = (const float*)d_in[6];
  const float* bk = (const float*)d_in[7];
  const float* Wo = (const float*)d_in[10];
  const float* bo = (const float*)d_in[11];
  float* out = (float*)d_out;

  const size_t HS = (size_t)BB * NH * SS * DK;    // 4M elements
  uint16_t* Qp = (uint16_t*)d_ws;                 // [B,H,S,DK] bf16, 8 MB
  uint16_t* Kp = Qp + HS;                         // 8 MB
  uint16_t* Qt = Kp + HS;                         // [B,H,DK,S] bf16 (V^T), 8 MB
  uint16_t* Xa = Qt + HS;                         // [B,S,D] bf16, 8 MB

  dim3 gg(32, 16), blk(256);
  gemm_xwt<0><<<gg, blk, 0, stream>>>(q, Wq, bq, Qp, Qt);
  gemm_xwt<0><<<gg, blk, 0, stream>>>(k, Wk, bk, Kp, nullptr);
  attn_kernel<<<dim3(16, 32), blk, 0, stream>>>(Qp, Qt, Kp, mask, Xa);
  gemm_xwt<1><<<gg, blk, 0, stream>>>(Xa, Wo, bo, out, nullptr);
}
